// SquadModel_28587302322491
// MI455X (gfx1250) — matrix operand order repaired and checked by reading, not compiled, not run on hardware
//
#include <hip/hip_runtime.h>

// ---------------------------------------------------------------------------
// Problem dims (compile-time, from the reference)
// ---------------------------------------------------------------------------
#define PP 400
#define QQ 30
#define BB 32
#define EE 300
#define HH 300
#define FF 300
#define AA 30
// padded dims (K multiples of 32, N multiples of 16)
#define E_P 320      // 300 -> 320
#define D_P 608      // 600 -> 608   (2E / 2H)
#define G_P 912      // 900 -> 912   (3H)
#define Q_P 32       // 30  -> 32

typedef __attribute__((ext_vector_type(16))) _Float16 v16h;
typedef __attribute__((ext_vector_type(8)))  _Float16 h8;
typedef __attribute__((ext_vector_type(8)))  float    v8f;

#define WMMA(a, b, c) \
  __builtin_amdgcn_wmma_f32_16x16x32_f16(false, (a), false, (b), (short)0, (c), false, false)

// A fragment (16x32 f16, M x K): lane = hi*16+li holds row (m0+li),
// halves j<8 -> K = k + hi*8 + j ; j>=8 -> K = k + 16 + hi*8 + (j-8)
__device__ __forceinline__ v16h frag_a(const _Float16* row, int k, int hi) {
  h8 lo = *(const h8*)(row + k + hi * 8);
  h8 hv = *(const h8*)(row + k + 16 + hi * 8);
  return __builtin_shufflevector(lo, hv, 0, 1, 2, 3, 4, 5, 6, 7, 8, 9, 10, 11, 12, 13, 14, 15);
}
// B fragment (32x16 f16, K x N) sourced from row-major W[n][k]: lane holds
// column n = n0+li, halves j<8 -> K = k + hi*8 + j ; j>=8 -> K = k + 16 + hi*8 + (j-8)
__device__ __forceinline__ v16h frag_b(const _Float16* row, int k, int hi) {
  h8 lo = *(const h8*)(row + k + hi * 8);
  h8 hv = *(const h8*)(row + k + 16 + hi * 8);
  return __builtin_shufflevector(lo, hv, 0, 1, 2, 3, 4, 5, 6, 7, 8, 9, 10, 11, 12, 13, 14, 15);
}

// ---------------------------------------------------------------------------
// Generic WMMA GEMM:  C[M,N] = act( A[M,K] * W[N,K]^T + bias )
// A: f16 padded rows (lda halves), W: f16 padded [Np, ldb], C: f32 [M, ldc]
// one 16x16 output tile per wave, 8 waves per block
// ---------------------------------------------------------------------------
__global__ void __launch_bounds__(256)
gemm_wmma(const _Float16* __restrict__ A, int lda,
          const _Float16* __restrict__ W, int ldb,
          const float* __restrict__ bias,
          float* __restrict__ C, int ldc,
          int ntiles, int total_tiles, int N, int ksteps, int do_relu) {
  int wid = threadIdx.x >> 5;
  int tile = blockIdx.x * 8 + wid;
  if (tile >= total_tiles) return;            // wave-uniform
  int lane = threadIdx.x & 31;
  int li = lane & 15, hi = lane >> 4;
  int m0 = (tile / ntiles) * 16;
  int n0 = (tile % ntiles) * 16;
  const _Float16* Arow = A + (size_t)(m0 + li) * lda;
  const _Float16* Wrow = W + (size_t)(n0 + li) * ldb;
  v8f acc = {};
  for (int ks = 0; ks < ksteps; ++ks) {
    int k = ks * 32;
    v16h af = frag_a(Arow, k, hi);
    v16h bf = frag_b(Wrow, k, hi);
    acc = WMMA(af, bf, acc);
  }
  int n = n0 + li;
  if (n < N) {
    float bv = bias ? bias[n] : 0.f;
#pragma unroll
    for (int i = 0; i < 8; ++i) {
      float v = acc[i] + bv;
      if (do_relu) v = fmaxf(v, 0.f);
      C[(size_t)(m0 + i + 8 * hi) * ldc + n] = v;
    }
  }
}

// ---------------------------------------------------------------------------
// scores[b,p,q] = sum_f ffp[p,b,f] * ffq[q,b,f]   (batched WMMA, N=Q padded 32)
// ffp16: [(p*32+b), 320] f16 ; ffq16: [(q*32+b), 320] f16 (rows q>=30 zero)
// ---------------------------------------------------------------------------
__global__ void __launch_bounds__(256)
scores_wmma(const _Float16* __restrict__ ffp, const _Float16* __restrict__ ffq,
            float* __restrict__ scores) {
  int wid = threadIdx.x >> 5;
  int tg = blockIdx.x * 8 + wid;              // 32 * 25 * 2 = 1600 tiles
  if (tg >= 1600) return;
  int b = tg / 50, rem = tg % 50;
  int mt = rem >> 1, nt = rem & 1;
  int lane = threadIdx.x & 31;
  int li = lane & 15, hi = lane >> 4;
  const _Float16* Arow = ffp + ((size_t)(mt * 16 + li) * BB + b) * E_P;
  const _Float16* Brow = ffq + ((size_t)(nt * 16 + li) * BB + b) * E_P;
  v8f acc = {};
  for (int ks = 0; ks < 10; ++ks) {
    int k = ks * 32;
    v16h af = frag_a(Arow, k, hi);
    v16h bf = frag_b(Brow, k, hi);
    acc = WMMA(af, bf, acc);
  }
  int q = nt * 16 + li;
  if (q < QQ) {
#pragma unroll
    for (int i = 0; i < 8; ++i)
      scores[(size_t)b * (PP * QQ) + (size_t)(mt * 16 + i + 8 * hi) * QQ + q] = acc[i];
  }
}

// softmax over q of (scores * p_mask * q_mask), in place. one thread per (b,p)
__global__ void softmax_mask(float* __restrict__ scores,
                             const float* __restrict__ p_mask,
                             const float* __restrict__ q_mask) {
  int idx = blockIdx.x * blockDim.x + threadIdx.x;
  if (idx >= BB * PP) return;
  int b = idx / PP, p = idx % PP;
  float* row = scores + (size_t)b * (PP * QQ) + (size_t)p * QQ;
  float pm = p_mask[p * BB + b];
  float s[QQ], mx = -1e30f;
  for (int q = 0; q < QQ; ++q) {
    s[q] = row[q] * pm * q_mask[q * BB + b];
    mx = fmaxf(mx, s[q]);
  }
  float sum = 0.f;
  for (int q = 0; q < QQ; ++q) { s[q] = __expf(s[q] - mx); sum += s[q]; }
  float inv = 1.f / sum;
  for (int q = 0; q < QQ; ++q) row[q] = s[q] * inv;
}

// q_align[p,b,e] = sum_q w[b,p,q] * qemb[q,b,e]; writes f16 into pstar cols 300..599
__global__ void q_align_fill(const float* __restrict__ weights,
                             const float* __restrict__ qemb,
                             _Float16* __restrict__ pstar) {
  int p = blockIdx.x >> 5, b = blockIdx.x & 31;
  __shared__ float w[QQ];
  if (threadIdx.x < QQ) w[threadIdx.x] = weights[(size_t)b * (PP * QQ) + (size_t)p * QQ + threadIdx.x];
  __syncthreads();
  for (int e = threadIdx.x; e < EE; e += blockDim.x) {
    float acc = 0.f;
#pragma unroll
    for (int q = 0; q < QQ; ++q) acc += w[q] * qemb[((size_t)q * BB + b) * EE + e];
    pstar[((size_t)p * BB + b) * D_P + EE + e] = (_Float16)acc;
  }
}

// ---------------------------------------------------------------------------
// GRU recurrence: one block per direction, 19 waves. h kept in LDS (f16+f32).
// Per step, wave w owns output columns j0=w*16..+15 and computes 6 WMMA tiles
// (2 M-tiles x {r,z,n} gates), then does the gate math and the state update.
// xp already contains x@Wih^T + b_ih.
// ---------------------------------------------------------------------------
__global__ void __launch_bounds__(608)
gru_layer(const float* __restrict__ xpF, const float* __restrict__ xpB,
          const _Float16* __restrict__ whh_base,   // + d*G_P*E_P
          const float* __restrict__ bhh_base,      // + d*900
          _Float16* __restrict__ out) {            // [T*B, 608], colOff = d*300
  const int d = blockIdx.x;
  const float* xp = d ? xpB : xpF;
  const _Float16* whh = whh_base + (size_t)d * G_P * E_P;
  const float* bhh = bhh_base + d * (3 * HH);
  const int colOff = d * HH;
  const int tid = threadIdx.x, lane = tid & 31, wid = tid >> 5;
  const int li = lane & 15, hi = lane >> 4;
  const int j0 = wid * 16;

  __shared__ __align__(16) _Float16 h16[BB * E_P];  // padded f16 state
  __shared__ float h32[BB * HH];                    // f32 state
  for (int i = tid; i < BB * E_P; i += blockDim.x) h16[i] = (_Float16)0.f;
  for (int i = tid; i < BB * HH; i += blockDim.x) h32[i] = 0.f;
  __syncthreads();

  const int jc = j0 + li;
  const bool jv = jc < HH;
  const _Float16* wr = whh + (size_t)(jc) * E_P;           // r-gate row
  const _Float16* wz = whh + (size_t)(HH + jc) * E_P;      // z-gate row
  const _Float16* wn = whh + (size_t)(2 * HH + jc) * E_P;  // n-gate row
  float br = 0.f, bz = 0.f, bn = 0.f;
  if (jv) { br = bhh[jc]; bz = bhh[HH + jc]; bn = bhh[2 * HH + jc]; }

  for (int step = 0; step < PP; ++step) {
    const int t = d ? (PP - 1 - step) : step;
    v8f ar0 = {}, az0 = {}, an0 = {}, ar1 = {}, az1 = {}, an1 = {};
    for (int ks = 0; ks < 10; ++ks) {   // K = 320 (300 + zero pad)
      int k = ks * 32;
      v16h a0 = frag_a(h16 + (size_t)li * E_P, k, hi);          // rows 0..15
      v16h a1 = frag_a(h16 + (size_t)(16 + li) * E_P, k, hi);   // rows 16..31
      v16h fr = frag_b(wr, k, hi);
      v16h fz = frag_b(wz, k, hi);
      v16h fn = frag_b(wn, k, hi);
      ar0 = WMMA(a0, fr, ar0);  ar1 = WMMA(a1, fr, ar1);
      az0 = WMMA(a0, fz, az0);  az1 = WMMA(a1, fz, az1);
      an0 = WMMA(a0, fn, an0);  an1 = WMMA(a1, fn, an1);
    }
    __syncthreads();   // everyone done reading h16 for this step
    if (jv) {
      const float* xrow = xp + (size_t)t * BB * (3 * HH);
#pragma unroll
      for (int half = 0; half < 2; ++half) {
        v8f Ar = half ? ar1 : ar0;
        v8f Az = half ? az1 : az0;
        v8f An = half ? an1 : an0;
#pragma unroll
        for (int i = 0; i < 8; ++i) {
          int b = half * 16 + i + 8 * hi;
          float xr = xrow[b * (3 * HH) + jc];
          float xz = xrow[b * (3 * HH) + HH + jc];
          float xn = xrow[b * (3 * HH) + 2 * HH + jc];
          float hr = Ar[i] + br, hz = Az[i] + bz, hn = An[i] + bn;
          float r = 1.f / (1.f + __expf(-(xr + hr)));
          float z = 1.f / (1.f + __expf(-(xz + hz)));
          float nn = tanhf(xn + r * hn);
          float hp = h32[b * HH + jc];
          float hnw = (1.f - z) * nn + z * hp;
          h32[b * HH + jc] = hnw;
          h16[b * E_P + jc] = (_Float16)hnw;
          out[((size_t)t * BB + b) * D_P + colOff + jc] = (_Float16)hnw;
        }
      }
    }
    __syncthreads();   // writes visible before next step's reads
  }
}

// span scoring: one wave per output (b, p*A+j); lanes reduce over F
__global__ void __launch_bounds__(256)
span_score(const float* __restrict__ stt, const float* __restrict__ endv,
           const float* __restrict__ wa, const int* __restrict__ p_lens,
           float* __restrict__ fin) {
  int wid = threadIdx.x >> 5;
  long oid = (long)blockIdx.x * 8 + wid;
  if (oid >= (long)BB * PP * AA) return;
  int b = (int)(oid / (PP * AA));
  int n = (int)(oid % (PP * AA));
  int p = n / AA, j = n % AA, pe = p + j;
  int lane = threadIdx.x & 31;
  const float* srow = stt + ((size_t)p * BB + b) * FF;
  const float* erow = (pe < PP) ? endv + ((size_t)pe * BB + b) * FF : nullptr;
  float acc = 0.f;
  for (int f = lane; f < FF; f += 32) {
    float v = srow[f] + (erow ? erow[f] : 0.f);
    acc += fmaxf(v, 0.f) * wa[f];
  }
#pragma unroll
  for (int off = 16; off; off >>= 1) acc += __shfl_xor(acc, off, 32);
  if (lane == 0) {
    float mask = (pe < p_lens[b]) ? 1.f : 0.f;
    fin[(size_t)b * (PP * AA) + n] = mask * acc;
  }
}

// log_softmax per batch row of 12000
__global__ void __launch_bounds__(256)
logsoftmax_row(const float* __restrict__ fin, float* __restrict__ outp) {
  const int NROW = PP * AA;
  int b = blockIdx.x;
  const float* row = fin + (size_t)b * NROW;
  float* orow = outp + (size_t)b * NROW;
  __shared__ float red[256];
  float mx = -1e30f;
  for (int i = threadIdx.x; i < NROW; i += 256) mx = fmaxf(mx, row[i]);
  red[threadIdx.x] = mx;
  __syncthreads();
  for (int s = 128; s; s >>= 1) {
    if (threadIdx.x < s) red[threadIdx.x] = fmaxf(red[threadIdx.x], red[threadIdx.x + s]);
    __syncthreads();
  }
  mx = red[0];
  __syncthreads();
  float sum = 0.f;
  for (int i = threadIdx.x; i < NROW; i += 256) sum += __expf(row[i] - mx);
  red[threadIdx.x] = sum;
  __syncthreads();
  for (int s = 128; s; s >>= 1) {
    if (threadIdx.x < s) red[threadIdx.x] += red[threadIdx.x + s];
    __syncthreads();
  }
  float ls = logf(red[0]);
  for (int i = threadIdx.x; i < NROW; i += 256) orow[i] = row[i] - mx - ls;
}

// ---------------------------------------------------------------------------
// helper kernels: f32 -> padded f16 conversion, gathers, zero-fill
// ---------------------------------------------------------------------------
__global__ void cvt_pad(const float* __restrict__ src, _Float16* __restrict__ dst,
                        int R, int C, int Cp, long total) {
  long idx = (long)blockIdx.x * 256 + threadIdx.x;
  if (idx >= total) return;
  int r = (int)(idx / Cp), c = (int)(idx % Cp);
  dst[idx] = (r < R && c < C) ? (_Float16)src[(size_t)r * C + c] : (_Float16)0.f;
}

__global__ void fill_zero_h(_Float16* __restrict__ p, long n) {
  long idx = (long)blockIdx.x * 256 + threadIdx.x;
  if (idx < n) p[idx] = (_Float16)0.f;
}

__global__ void gather_p(const int* __restrict__ ids, const float* __restrict__ embed,
                         _Float16* __restrict__ pstar) {
  long idx = (long)blockIdx.x * 256 + threadIdx.x;
  if (idx >= (long)PP * BB * EE) return;
  int e = (int)(idx % EE);
  long pb = idx / EE;                         // p*B + b
  int tok = ids[pb];
  pstar[pb * D_P + e] = (_Float16)embed[(size_t)tok * EE + e];
}

__global__ void gather_q(const int* __restrict__ ids, const float* __restrict__ embed,
                         float* __restrict__ qemb) {
  long idx = (long)blockIdx.x * 256 + threadIdx.x;
  if (idx >= (long)QQ * BB * EE) return;
  int e = (int)(idx % EE);
  long qb = idx / EE;
  qemb[idx] = embed[(size_t)ids[qb] * EE + e];
}

// ---------------------------------------------------------------------------
// host-side orchestration
// ---------------------------------------------------------------------------
static inline int blks(long n) { return (int)((n + 255) / 256); }

extern "C" void kernel_launch(void* const* d_in, const int* in_sizes, int n_in,
                              void* d_out, int out_size, void* d_ws, size_t ws_size,
                              hipStream_t stream) {
  (void)in_sizes; (void)n_in; (void)out_size;
  const int*   p_ids  = (const int*)d_in[0];
  const int*   q_ids  = (const int*)d_in[1];
  const float* p_mask = (const float*)d_in[2];
  const float* q_mask = (const float*)d_in[3];
  const int*   p_lens = (const int*)d_in[4];
  const float* embed  = (const float*)d_in[6];
  const float* w_align = (const float*)d_in[7];
  const float* b_align = (const float*)d_in[8];
  const float* gwih = (const float*)d_in[9];    // [2,2,900,600]
  const float* gwhh = (const float*)d_in[10];   // [2,2,900,300]
  const float* gbih = (const float*)d_in[11];   // [2,2,900]
  const float* gbhh = (const float*)d_in[12];   // [2,2,900]
  const float* w_stt = (const float*)d_in[13];
  const float* b_stt = (const float*)d_in[14];
  const float* w_end = (const float*)d_in[15];
  const float* b_end = (const float*)d_in[16];
  const float* w_a   = (const float*)d_in[17];
  float* outp = (float*)d_out;

  // bump allocator over workspace (~190 MB)
  char* base = (char*)d_ws;
  size_t off = 0;
  auto alloc = [&](size_t bytes) -> char* {
    char* r = base + off;
    off = (off + bytes + 255) & ~(size_t)255;
    return r;
  };
  _Float16* Wa16   = (_Float16*)alloc((size_t)304 * E_P * 2);
  _Float16* Wih16  = (_Float16*)alloc((size_t)4 * G_P * D_P * 2);
  _Float16* Whh16  = (_Float16*)alloc((size_t)4 * G_P * E_P * 2);
  _Float16* Wstt16 = (_Float16*)alloc((size_t)304 * D_P * 2);
  _Float16* Wend16 = (_Float16*)alloc((size_t)304 * D_P * 2);
  float*    qembf  = (float*)alloc((size_t)QQ * BB * EE * 4);
  _Float16* qemb16 = (_Float16*)alloc((size_t)QQ * BB * E_P * 2);
  _Float16* pstar16 = (_Float16*)alloc((size_t)PP * BB * D_P * 2);
  float*    fff    = (float*)alloc((size_t)PP * BB * FF * 4);
  float*    ffqf   = (float*)alloc((size_t)QQ * BB * FF * 4);
  _Float16* ffp16  = (_Float16*)alloc((size_t)PP * BB * E_P * 2);
  _Float16* ffq16  = (_Float16*)alloc((size_t)Q_P * BB * E_P * 2);
  float*    scr    = (float*)alloc((size_t)BB * PP * QQ * 4);
  float*    xpF    = (float*)alloc((size_t)PP * BB * 3 * HH * 4);
  float*    xpB    = (float*)alloc((size_t)PP * BB * 3 * HH * 4);
  _Float16* h1o    = (_Float16*)alloc((size_t)PP * BB * D_P * 2);
  _Float16* h2o    = (_Float16*)alloc((size_t)PP * BB * D_P * 2);
  (void)ws_size;
  // stt/end/final alias the (then-dead) xp buffers
  float* sttf = xpF;
  float* endf = xpF + (size_t)PP * BB * FF;
  float* fin  = xpB;

  // 1) zero-fill padded f16 activation buffers (pads must be exact zeros)
  fill_zero_h<<<blks((long)PP * BB * D_P), 256, 0, stream>>>(pstar16, (long)PP * BB * D_P);
  fill_zero_h<<<blks((long)PP * BB * D_P), 256, 0, stream>>>(h1o, (long)PP * BB * D_P);
  fill_zero_h<<<blks((long)PP * BB * D_P), 256, 0, stream>>>(h2o, (long)PP * BB * D_P);

  // 2) weight conversion to padded f16
  cvt_pad<<<blks((long)304 * E_P), 256, 0, stream>>>(w_align, Wa16, FF, EE, E_P, (long)304 * E_P);
  for (int s = 0; s < 4; ++s) {
    cvt_pad<<<blks((long)G_P * D_P), 256, 0, stream>>>(
        gwih + (size_t)s * 900 * 600, Wih16 + (size_t)s * G_P * D_P, 900, 600, D_P, (long)G_P * D_P);
    cvt_pad<<<blks((long)G_P * E_P), 256, 0, stream>>>(
        gwhh + (size_t)s * 900 * 300, Whh16 + (size_t)s * G_P * E_P, 900, 300, E_P, (long)G_P * E_P);
  }
  cvt_pad<<<blks((long)304 * D_P), 256, 0, stream>>>(w_stt, Wstt16, FF, 600, D_P, (long)304 * D_P);
  cvt_pad<<<blks((long)304 * D_P), 256, 0, stream>>>(w_end, Wend16, FF, 600, D_P, (long)304 * D_P);

  // 3) embedding gathers
  gather_p<<<blks((long)PP * BB * EE), 256, 0, stream>>>(p_ids, embed, pstar16);
  gather_q<<<blks((long)QQ * BB * EE), 256, 0, stream>>>(q_ids, embed, qembf);
  cvt_pad<<<blks((long)QQ * BB * E_P), 256, 0, stream>>>(qembf, qemb16, QQ * BB, EE, E_P, (long)QQ * BB * E_P);

  // 4) alignment features: ff = relu(emb @ w_align^T + b)
  gemm_wmma<<<(800 * 19 + 7) / 8, 256, 0, stream>>>(pstar16, D_P, Wa16, E_P, b_align,
                                                    fff, FF, 19, 800 * 19, FF, 10, 1);
  gemm_wmma<<<(60 * 19 + 7) / 8, 256, 0, stream>>>(qemb16, E_P, Wa16, E_P, b_align,
                                                   ffqf, FF, 19, 60 * 19, FF, 10, 1);
  cvt_pad<<<blks((long)PP * BB * E_P), 256, 0, stream>>>(fff, ffp16, PP * BB, FF, E_P, (long)PP * BB * E_P);
  cvt_pad<<<blks((long)Q_P * BB * E_P), 256, 0, stream>>>(ffqf, ffq16, QQ * BB, FF, E_P, (long)Q_P * BB * E_P);

  // 5) attention scores + softmax + q_align (writes pstar cols 300..599)
  scores_wmma<<<200, 256, 0, stream>>>(ffp16, ffq16, scr);
  softmax_mask<<<blks(BB * PP), 256, 0, stream>>>(scr, p_mask, q_mask);
  q_align_fill<<<PP * BB, 256, 0, stream>>>(scr, qembf, pstar16);

  // 6) BiGRU layers: xp = x @ Wih^T + b_ih (WMMA), then recurrence (WMMA)
  const int xp_tiles = 800 * 57;
  // layer 1
  gemm_wmma<<<(xp_tiles + 7) / 8, 256, 0, stream>>>(pstar16, D_P, Wih16 + (size_t)0 * G_P * D_P, D_P,
                                                    gbih + 0, xpF, 3 * HH, 57, xp_tiles, 3 * HH, 19, 0);
  gemm_wmma<<<(xp_tiles + 7) / 8, 256, 0, stream>>>(pstar16, D_P, Wih16 + (size_t)1 * G_P * D_P, D_P,
                                                    gbih + 900, xpB, 3 * HH, 57, xp_tiles, 3 * HH, 19, 0);
  gru_layer<<<2, 608, 0, stream>>>(xpF, xpB, Whh16 + (size_t)0 * G_P * E_P, gbhh + 0, h1o);
  // layer 2
  gemm_wmma<<<(xp_tiles + 7) / 8, 256, 0, stream>>>(h1o, D_P, Wih16 + (size_t)2 * G_P * D_P, D_P,
                                                    gbih + 1800, xpF, 3 * HH, 57, xp_tiles, 3 * HH, 19, 0);
  gemm_wmma<<<(xp_tiles + 7) / 8, 256, 0, stream>>>(h1o, D_P, Wih16 + (size_t)3 * G_P * D_P, D_P,
                                                    gbih + 2700, xpB, 3 * HH, 57, xp_tiles, 3 * HH, 19, 0);
  gru_layer<<<2, 608, 0, stream>>>(xpF, xpB, Whh16 + (size_t)2 * G_P * E_P, gbhh + 1800, h2o);

  // 7) start/end projections with relu (aliased onto dead xpF)
  gemm_wmma<<<(800 * 19 + 7) / 8, 256, 0, stream>>>(h2o, D_P, Wstt16, D_P, b_stt,
                                                    sttf, FF, 19, 800 * 19, FF, 19, 1);
  gemm_wmma<<<(800 * 19 + 7) / 8, 256, 0, stream>>>(h2o, D_P, Wend16, D_P, b_end,
                                                    endf, FF, 19, 800 * 19, FF, 19, 1);

  // 8) span scoring + log_softmax
  span_score<<<(BB * PP * AA + 7) / 8, 256, 0, stream>>>(sttf, endf, w_a, p_lens, fin);
  logsoftmax_row<<<BB, 256, 0, stream>>>(fin, outp);
}
